// ElectrostaticFieldInteractionBlock_18502719111824
// MI455X (gfx1250) — matrix operand order repaired and checked by reading, not compiled, not run on hardware
//
#include <hip/hip_runtime.h>

typedef __attribute__((ext_vector_type(16))) _Float16 v16h;
typedef __attribute__((ext_vector_type(8)))  float    v8f;

#define N_NODES 65536
#define TILES   (N_NODES / 16)   // 4096 node tiles of 16
#define WPB     8                // waves per block
#define SZH     8192             // f16 elems per wave LDS region: S(16x128) + V(3x16x128)

// d_ws layout (f16 elements):
//   [0, 49152)          WsFrag: 24 ntiles x 4 kt x 32 lanes x 16
//   [49152, 114688)     WvFrag: 32 ntiles x 4 kt x 32 lanes x 16
#define WS_S_ELEMS 49152

// ---------------------------------------------------------------------------
// Prep: fold W_up, tp_w, W_l and all path normalizations into two combined
// weight matrices, stored pre-swizzled in the WMMA B-matrix lane layout.
// ---------------------------------------------------------------------------
__global__ __launch_bounds__(256) void prep_weights(
    const float* __restrict__ Wup0, const float* __restrict__ Wup1,
    const float* __restrict__ tpw,
    const float* __restrict__ Wl0,  const float* __restrict__ Wl1,
    const float* __restrict__ Wl2,
    _Float16* __restrict__ wsOut)
{
    int t = blockIdx.x * 256 + threadIdx.x;
    const float INV_UP = 0.08838834764831845f;   // 1/sqrt(128)
    const float A2 = 0.70710678118654752f;       // 1/sqrt(2)
    const float A3 = 0.57735026918962576f;       // 1/sqrt(3)
    const float I0 = 0.0625f;                    // 1/sqrt(256)
    const float I1 = 0.05103103630798288f;       // 1/sqrt(384)
    const float I2 = 0.0625f;                    // 1/sqrt(256)

    if (t < 128 * 384) {
        // Ws: columns [p0 | p1 | p2]
        int k = t / 384, col = t - k * 384;
        int p = col >> 7, c = col & 127;
        const float* wl; const float* tw; float sc;
        if (p == 0)      { wl = Wl0;  tw = tpw;         sc = A2 * I0; }
        else if (p == 1) { wl = Wl1;  tw = tpw + 128;   sc = A3 * I1; }
        else             { wl = Wl2;  tw = tpw + 256;   sc = A2 * I2; }
        float acc = 0.f;
        for (int u = 0; u < 128; ++u)
            acc += Wup0[k * 128 + u] * tw[u] * wl[u * 128 + c];
        acc *= sc * INV_UP;
        int nt = col >> 4, kt = k >> 5, rem = k & 31;
        int lane = (col & 15) | ((rem & 8) << 1);
        int e = (rem & 7) | ((rem & 16) >> 1);
        wsOut[(nt * 4 + kt) * 512 + lane * 16 + e] = (_Float16)acc;
    } else if (t < 128 * 384 + 128 * 512) {
        // Wv: columns [p3 | p4 | p5 | p6]
        int t2 = t - 128 * 384;
        int k = t2 / 512, col = t2 & 511;
        int p = col >> 7, c = col & 127;
        const float* wl; const float* tw; float sc;
        if (p == 0)      { wl = Wl1 + 128 * 128; tw = tpw + 3 * 128; sc = A3 * I1; }
        else if (p == 1) { wl = Wl0 + 128 * 128; tw = tpw + 4 * 128; sc = A2 * I0; }
        else if (p == 2) { wl = Wl2 + 128 * 128; tw = tpw + 5 * 128; sc = A2 * I2; }
        else             { wl = Wl1 + 256 * 128; tw = tpw + 6 * 128; sc = A3 * I1; }
        float acc = 0.f;
        for (int u = 0; u < 128; ++u)
            acc += Wup1[k * 128 + u] * tw[u] * wl[u * 128 + c];
        acc *= sc * INV_UP;
        int nt = col >> 4, kt = k >> 5, rem = k & 31;
        int lane = (col & 15) | ((rem & 8) << 1);
        int e = (rem & 7) | ((rem & 16) >> 1);
        wsOut[WS_S_ELEMS + (nt * 4 + kt) * 512 + lane * 16 + e] = (_Float16)acc;
    }
}

// ---------------------------------------------------------------------------
// Fused GEMM (WMMA f16 -> f32) + w3j epilogue.
// One wave = one 16-node tile; loops over 8 v-slices of 16 output channels.
// ---------------------------------------------------------------------------
__global__ __launch_bounds__(256, 3) void fused_kernel(
    const float* __restrict__ nf,     // node_feats (N, 512)
    const float* __restrict__ mp,     // multipole_feats (N, 9)
    const _Float16* __restrict__ wfrag,
    float* __restrict__ out)          // (N, 128, 9)
{
    __shared__ _Float16 ldsA[WPB * SZH];   // per-wave: S[16][128] then V[3][16][128]
    __shared__ float    ldsMP[WPB * 144];  // per-wave: MP[16][9]

    const int wv   = threadIdx.x >> 5;
    const int lane = threadIdx.x & 31;
    const int tile = blockIdx.x * WPB + wv;
    const int nb   = tile * 16;

    _Float16* Sb  = ldsA + wv * SZH;
    _Float16* Vb  = Sb + 2048;
    float*    MPb = ldsMP + wv * 144;

    // ---- Stage tile: f32 -> f16, deinterleave v components -----------------
    const float4* src = (const float4*)(nf + (size_t)nb * 512);
    for (int i = lane; i < 2048; i += 32) {
        float4 x = src[i];
        int node = i >> 7;
        int col  = (i & 127) << 2;
        if (col < 128) {
            Sb[node * 128 + col + 0] = (_Float16)x.x;
            Sb[node * 128 + col + 1] = (_Float16)x.y;
            Sb[node * 128 + col + 2] = (_Float16)x.z;
            Sb[node * 128 + col + 3] = (_Float16)x.w;
        } else {
            int c = col - 128;
            float vals[4] = {x.x, x.y, x.z, x.w};
            #pragma unroll
            for (int j = 0; j < 4; ++j) {
                int cc = c + j, u = cc / 3, a = cc - u * 3;
                Vb[a * 2048 + node * 128 + u] = (_Float16)vals[j];
            }
        }
    }
    for (int i = lane; i < 144; i += 32) {
        int node = i / 9, j = i - node * 9;
        MPb[i] = mp[(size_t)(nb + node) * 9 + j];
    }

    const int m    = lane & 15;
    const int koff = (lane & 16) ? 8 : 0;
    const _Float16* wsF = wfrag;
    const _Float16* wvF = wfrag + WS_S_ELEMS;

    const float R3  = 0.57735026918962576f;   // 1/sqrt(3)
    const float R5  = 0.44721359549995794f;   // 1/sqrt(5)
    const float S10 = 0.31622776601683794f;   // 1/sqrt(10)
    const float D30 = 0.18257418583505536f;   // 1/sqrt(30)
    const float D60 = 0.36514837167011072f;   // 2/sqrt(30)

    #pragma clang loop unroll(disable)
    for (int vs = 0; vs < 8; ++vs) {
        // Launder a zero offset so LICM cannot hoist the (vs-invariant)
        // A-fragment / multipole LDS loads out of this loop: keeping all four
        // k-steps of A live across the loop costs ~128 VGPRs and caused
        // scratch spills. Adding an opaque *integer* (not laundering the
        // pointers themselves) preserves addrspace(3) inference, so these
        // stay ds_load_b128 instead of decaying to flat_load_b128.
        int zofs = 0;
        asm volatile("" : "+v"(zofs));
        const _Float16* Sv  = Sb  + zofs;
        const _Float16* Vv  = Vb  + zofs;
        const float*    MPv = MPb + zofs;

        v8f acc[15] = {};   // [0..2]=ys p0..p2, [3..5]=p3 a0..2, [6..8]=p4, [9..11]=p5, [12..14]=p6

        #pragma unroll
        for (int kt = 0; kt < 4; ++kt) {
            // A fragments from LDS (16-bit A 16x32 layout)
            v16h As, Av0, Av1, Av2;
            {
                const _Float16* p0 = Sv + m * 128 + kt * 32 + koff;
                ((uint4*)&As)[0] = *(const uint4*)(p0);
                ((uint4*)&As)[1] = *(const uint4*)(p0 + 16);
                const _Float16* p1 = Vv + 0 * 2048 + m * 128 + kt * 32 + koff;
                ((uint4*)&Av0)[0] = *(const uint4*)(p1);
                ((uint4*)&Av0)[1] = *(const uint4*)(p1 + 16);
                const _Float16* p2 = Vv + 1 * 2048 + m * 128 + kt * 32 + koff;
                ((uint4*)&Av1)[0] = *(const uint4*)(p2);
                ((uint4*)&Av1)[1] = *(const uint4*)(p2 + 16);
                const _Float16* p3 = Vv + 2 * 2048 + m * 128 + kt * 32 + koff;
                ((uint4*)&Av2)[0] = *(const uint4*)(p3);
                ((uint4*)&Av2)[1] = *(const uint4*)(p3 + 16);
            }
            // B fragments from global (pre-swizzled, L2-resident)
            auto ldB = [&](const _Float16* base, int frag) -> v16h {
                v16h b;
                const uint4* q = (const uint4*)(base + (size_t)frag * 512 + lane * 16);
                ((uint4*)&b)[0] = q[0];
                ((uint4*)&b)[1] = q[1];
                return b;
            };
            v16h Bs0 = ldB(wsF, (0  + vs) * 4 + kt);
            v16h Bs1 = ldB(wsF, (8  + vs) * 4 + kt);
            v16h Bs2 = ldB(wsF, (16 + vs) * 4 + kt);
            v16h Bv3 = ldB(wvF, (0  + vs) * 4 + kt);
            v16h Bv4 = ldB(wvF, (8  + vs) * 4 + kt);
            v16h Bv5 = ldB(wvF, (16 + vs) * 4 + kt);
            v16h Bv6 = ldB(wvF, (24 + vs) * 4 + kt);

            acc[0]  = __builtin_amdgcn_wmma_f32_16x16x32_f16(false, As,  false, Bs0, (short)0, acc[0],  false, false);
            acc[1]  = __builtin_amdgcn_wmma_f32_16x16x32_f16(false, As,  false, Bs1, (short)0, acc[1],  false, false);
            acc[2]  = __builtin_amdgcn_wmma_f32_16x16x32_f16(false, As,  false, Bs2, (short)0, acc[2],  false, false);
            acc[3]  = __builtin_amdgcn_wmma_f32_16x16x32_f16(false, Av0, false, Bv3, (short)0, acc[3],  false, false);
            acc[4]  = __builtin_amdgcn_wmma_f32_16x16x32_f16(false, Av1, false, Bv3, (short)0, acc[4],  false, false);
            acc[5]  = __builtin_amdgcn_wmma_f32_16x16x32_f16(false, Av2, false, Bv3, (short)0, acc[5],  false, false);
            acc[6]  = __builtin_amdgcn_wmma_f32_16x16x32_f16(false, Av0, false, Bv4, (short)0, acc[6],  false, false);
            acc[7]  = __builtin_amdgcn_wmma_f32_16x16x32_f16(false, Av1, false, Bv4, (short)0, acc[7],  false, false);
            acc[8]  = __builtin_amdgcn_wmma_f32_16x16x32_f16(false, Av2, false, Bv4, (short)0, acc[8],  false, false);
            acc[9]  = __builtin_amdgcn_wmma_f32_16x16x32_f16(false, Av0, false, Bv5, (short)0, acc[9],  false, false);
            acc[10] = __builtin_amdgcn_wmma_f32_16x16x32_f16(false, Av1, false, Bv5, (short)0, acc[10], false, false);
            acc[11] = __builtin_amdgcn_wmma_f32_16x16x32_f16(false, Av2, false, Bv5, (short)0, acc[11], false, false);
            acc[12] = __builtin_amdgcn_wmma_f32_16x16x32_f16(false, Av0, false, Bv6, (short)0, acc[12], false, false);
            acc[13] = __builtin_amdgcn_wmma_f32_16x16x32_f16(false, Av1, false, Bv6, (short)0, acc[13], false, false);
            acc[14] = __builtin_amdgcn_wmma_f32_16x16x32_f16(false, Av2, false, Bv6, (short)0, acc[14], false, false);
        }

        // ---- w3j epilogue: C/D layout -> lane holds (M = r + 8*(lane>=16), N = lane&15)
        const int v = (vs << 4) | m;
        #pragma unroll
        for (int r = 0; r < 8; ++r) {
            int node = r + ((lane >> 4) << 3);
            const float* f = MPv + node * 9;
            float f0  = f[0];
            float f10 = f[1], f11 = f[2], f12 = f[3];
            float f20 = f[4], f21 = f[5], f22 = f[6], f23 = f[7], f24 = f[8];

            float ys0 = acc[0][r],  ys1 = acc[1][r],  ys2 = acc[2][r];
            float y30 = acc[3][r],  y31 = acc[4][r],  y32 = acc[5][r];
            float y40 = acc[6][r],  y41 = acc[7][r],  y42 = acc[8][r];
            float y50 = acc[9][r],  y51 = acc[10][r], y52 = acc[11][r];
            float y60 = acc[12][r], y61 = acc[13][r], y62 = acc[14][r];

            float o0 = ys0 * f0 + R3 * (f10 * y40 + f11 * y41 + f12 * y42);
            float o1 = R3 * (ys1 * f10 + y30 * f0)
                     + y60 * (-D30 * f22 - S10 * f24) + y61 * (S10 * f21) + y62 * (S10 * f20);
            float o2 = R3 * (ys1 * f11 + y31 * f0)
                     + y60 * (S10 * f21) + y61 * (D60 * f22) + y62 * (S10 * f23);
            float o3 = R3 * (ys1 * f12 + y32 * f0)
                     + y60 * (S10 * f20) + y61 * (S10 * f23) + y62 * (-D30 * f22 + S10 * f24);
            float o4 = R5 * ys2 * f20 + S10 * (y50 * f12 + y52 * f10);
            float o5 = R5 * ys2 * f21 + S10 * (y50 * f11 + y51 * f10);
            float o6 = R5 * ys2 * f22 - D30 * (y50 * f10 + y52 * f12) + D60 * (y51 * f11);
            float o7 = R5 * ys2 * f23 + S10 * (y51 * f12 + y52 * f11);
            float o8 = R5 * ys2 * f24 + S10 * (y52 * f12 - y50 * f10);

            float* op = out + ((size_t)(nb + node) * 128 + v) * 9;
            op[0] = o0; op[1] = o1; op[2] = o2; op[3] = o3; op[4] = o4;
            op[5] = o5; op[6] = o6; op[7] = o7; op[8] = o8;
        }
    }
}

extern "C" void kernel_launch(void* const* d_in, const int* in_sizes, int n_in,
                              void* d_out, int out_size, void* d_ws, size_t ws_size,
                              hipStream_t stream) {
    const float* nf   = (const float*)d_in[0];
    const float* mpf  = (const float*)d_in[1];
    const float* Wup0 = (const float*)d_in[2];
    const float* Wup1 = (const float*)d_in[3];
    const float* tpw  = (const float*)d_in[4];
    const float* Wl0  = (const float*)d_in[5];
    const float* Wl1  = (const float*)d_in[6];
    const float* Wl2  = (const float*)d_in[7];
    _Float16* wfrag   = (_Float16*)d_ws;   // needs 229376 bytes

    int totalW = 128 * 384 + 128 * 512;    // 114688 combined-weight elements
    prep_weights<<<(totalW + 255) / 256, 256, 0, stream>>>(Wup0, Wup1, tpw, Wl0, Wl1, Wl2, wfrag);
    fused_kernel<<<TILES / WPB, 256, 0, stream>>>(nf, mpf, wfrag, (float*)d_out);
}
